// Transformer_70274254897751
// MI455X (gfx1250) — matrix operand order repaired and checked by reading, not compiled, not run on hardware
//
#include <hip/hip_runtime.h>
#include <math.h>

typedef __attribute__((ext_vector_type(16))) __bf16 v16bf;
typedef __attribute__((ext_vector_type(8)))  float  v8f;

#define CDIV(a, b) (((a) + (b) - 1) / (b))

// ---------------- WMMA fragment index maps (bf16 16x16x32) ----------------
// A (16x32, MxK): lanes 0-15 M=0..15 K in {0..7,16..23}; lanes16-31 K in {8..15,24..31}
__device__ __forceinline__ int koffA(int lane, int e) {
    int v = e >> 1, h = e & 1;
    return ((v < 4) ? 0 : 16) + ((lane & 16) ? 8 : 0) + ((v & 3) << 1) + h;
}
// B (32x16, KxN): lanes 0-15 N=0..15 K in {0..7,16..23}; lanes16-31 K in {8..15,24..31}
__device__ __forceinline__ int koffB(int lane, int e) {
    return ((e < 8) ? 0 : 16) + ((lane & 16) ? 8 : 0) + (e & 7);
}

// Pack W[128][nCols] (row-major fp32) into per-wave B fragments:
// layout [tile][kb][lane][16] bf16
__global__ void pack_weights_kernel(const float* __restrict__ W, __bf16* __restrict__ out,
                                    int nCols, int nTiles) {
    int idx = blockIdx.x * blockDim.x + threadIdx.x;
    int total = nTiles * 4 * 32 * 16;
    if (idx >= total) return;
    int e    = idx & 15;
    int lane = (idx >> 4) & 31;
    int kb   = (idx >> 9) & 3;
    int tile = idx >> 11;
    int col  = tile * 16 + (lane & 15);
    int k    = kb * 32 + koffB(lane, e);
    float val = (col < nCols) ? W[k * nCols + col] : 0.0f;
    out[idx] = (__bf16)val;
}

// D = A @ W_mat + b_mat, A [N,128] fp32, one matrix per blockIdx.y.
// Block = 64 rows; wave w owns col-tile w: preloads 4 B k-fragments (one
// load clause), then 4 row-tiles x 4 back-to-back WMMAs reusing B.
__global__ void gemm_wmma_v2(const float* __restrict__ A, const __bf16* __restrict__ packed,
                             float* o0, float* o1, float* o2, float* o3,
                             const float* b0, const float* b1,
                             const float* b2, const float* b3,
                             int N, int nCols, int outStride, int tilesPerMat) {
    __shared__ float lds[64 * 128];
    const int mat  = blockIdx.y;
    const int row0 = blockIdx.x * 64;
    const int tid  = threadIdx.x;

    for (int i = tid; i < 64 * 128; i += 256) {
        int r = i >> 7, c = i & 127;
        int row = row0 + r;
        lds[i] = (row < N) ? A[(size_t)row * 128 + c] : 0.0f;
    }
    __syncthreads();

    const int wave = tid >> 5, lane = tid & 31;
    if (wave >= tilesPerMat) return;

    float*       outp  = (mat == 0) ? o0 : (mat == 1) ? o1 : (mat == 2) ? o2 : o3;
    const float* biasp = (mat == 0) ? b0 : (mat == 1) ? b1 : (mat == 2) ? b2 : b3;

    const __bf16* pb = packed + ((size_t)mat * tilesPerMat + wave) * (4 * 32 * 16);
    v16bf bF[4];
#pragma unroll
    for (int kb = 0; kb < 4; ++kb)
        bF[kb] = *(const v16bf*)(pb + kb * 512 + lane * 16);

    const int col  = wave * 16 + (lane & 15);
    const int m    = lane & 15;
    const int rAdd = (lane & 16) ? 8 : 0;
    const float bv = (col < nCols) ? biasp[col] : 0.0f;

#pragma unroll
    for (int rt = 0; rt < 4; ++rt) {
        const float* at = lds + (rt * 16 + m) * 128;
        v16bf aF[4];
#pragma unroll
        for (int kb = 0; kb < 4; ++kb)
#pragma unroll
            for (int e = 0; e < 16; ++e)
                aF[kb][e] = (__bf16)at[kb * 32 + koffA(lane, e)];

        v8f acc = {};
        acc = __builtin_amdgcn_wmma_f32_16x16x32_bf16(false, aF[0], false, bF[0], (short)0, acc, false, false);
        acc = __builtin_amdgcn_wmma_f32_16x16x32_bf16(false, aF[1], false, bF[1], (short)0, acc, false, false);
        acc = __builtin_amdgcn_wmma_f32_16x16x32_bf16(false, aF[2], false, bF[2], (short)0, acc, false, false);
        acc = __builtin_amdgcn_wmma_f32_16x16x32_bf16(false, aF[3], false, bF[3], (short)0, acc, false, false);

        if (col < nCols) {
#pragma unroll
            for (int r = 0; r < 8; ++r) {
                int row = row0 + rt * 16 + r + rAdd;
                if (row < N) outp[(size_t)row * outStride + col] = acc[r] + bv;
            }
        }
    }
}

// ---------------- layer 1 linears (K=15, too small for WMMA) ----------------
__global__ void lin15_qkv(const float* __restrict__ bf,
                          const float* Wq, const float* bq,
                          const float* Wk, const float* bk,
                          const float* Wv, const float* bv,
                          float* Q, float* K, float* V, int N) {
    int idx = blockIdx.x * blockDim.x + threadIdx.x;
    if (idx >= N * 128) return;
    int n = idx >> 7, c = idx & 127;
    float q = bq[c], k = bk[c], v = bv[c];
#pragma unroll
    for (int i = 0; i < 15; ++i) {
        float x = bf[n * 15 + i];
        q = fmaf(x, Wq[i * 128 + c], q);
        k = fmaf(x, Wk[i * 128 + c], k);
        v = fmaf(x, Wv[i * 128 + c], v);
    }
    Q[idx] = q; K[idx] = k; V[idx] = v;
}

// ---------------- edge kernels ----------------
__device__ __forceinline__ void atomicMaxF(float* addr, float val) {
    if (val >= 0.0f) atomicMax((int*)addr, __float_as_int(val));
    else             atomicMin((unsigned int*)addr, __float_as_uint(val));
}

// pass 1 (C=128): alpha = q[dst].(k[src]+w*We)/sqrt(C); atomic max per dst
__global__ void edge_alpha128(const int* __restrict__ ei, const float* __restrict__ Q,
                              const float* __restrict__ K, const float* __restrict__ We,
                              const float* __restrict__ w, float* __restrict__ alpha,
                              float* __restrict__ mbuf, int E, float scale) {
    int warp = (blockIdx.x * blockDim.x + threadIdx.x) >> 5;
    int lane = threadIdx.x & 31;
    if (warp >= E) return;
    int src = ei[warp], dst = ei[E + warp];
    float we = w[warp];
    float4 q4 = *(const float4*)(Q + (size_t)dst * 128 + lane * 4);
    float4 k4 = *(const float4*)(K + (size_t)src * 128 + lane * 4);
    float4 e4 = *(const float4*)(We + lane * 4);
    float p = q4.x * fmaf(we, e4.x, k4.x) + q4.y * fmaf(we, e4.y, k4.y)
            + q4.z * fmaf(we, e4.z, k4.z) + q4.w * fmaf(we, e4.w, k4.w);
#pragma unroll
    for (int off = 16; off >= 1; off >>= 1) p += __shfl_xor(p, off, 32);
    if (lane == 0) {
        float a = p * scale;
        alpha[warp] = a;
        atomicMaxF(&mbuf[dst], a);
    }
}

// pass 2: p = exp(alpha - m[dst]); s[dst] += p
__global__ void edge_exp(const int* __restrict__ ei, float* __restrict__ alphaP,
                         const float* __restrict__ mbuf, float* __restrict__ sbuf, int E) {
    int e = blockIdx.x * blockDim.x + threadIdx.x;
    if (e >= E) return;
    int dst = ei[E + e];
    float p = __expf(alphaP[e] - mbuf[dst]);
    alphaP[e] = p;
    atomicAdd(&sbuf[dst], p);
}

// pass 3 (C=128): a = p/(s+eps); out[dst] += (v[src]+w*We)*a; store a
__global__ void edge_msg128(const int* __restrict__ ei, const float* __restrict__ V,
                            const float* __restrict__ We, const float* __restrict__ w,
                            const float* __restrict__ alphaP, const float* __restrict__ sbuf,
                            float* __restrict__ aOut, float* __restrict__ out, int E) {
    int warp = (blockIdx.x * blockDim.x + threadIdx.x) >> 5;
    int lane = threadIdx.x & 31;
    if (warp >= E) return;
    int src = ei[warp], dst = ei[E + warp];
    float we = w[warp];
    float a = alphaP[warp] / (sbuf[dst] + 1e-16f);
    if (lane == 0) aOut[warp] = a;
    float4 v4 = *(const float4*)(V + (size_t)src * 128 + lane * 4);
    float4 e4 = *(const float4*)(We + lane * 4);
    float* o = out + (size_t)dst * 128 + lane * 4;
    atomicAdd(o + 0, fmaf(we, e4.x, v4.x) * a);
    atomicAdd(o + 1, fmaf(we, e4.y, v4.y) * a);
    atomicAdd(o + 2, fmaf(we, e4.z, v4.z) * a);
    atomicAdd(o + 3, fmaf(we, e4.w, v4.w) * a);
}

// C=5 variants (layer 3), thread-per-edge
__global__ void edge_alpha5(const int* __restrict__ ei, const float* __restrict__ Q,
                            const float* __restrict__ K, const float* __restrict__ We,
                            const float* __restrict__ w, float* __restrict__ alpha,
                            float* __restrict__ mbuf, int E, float scale) {
    int e = blockIdx.x * blockDim.x + threadIdx.x;
    if (e >= E) return;
    int src = ei[e], dst = ei[E + e];
    float we = w[e];
    float p = 0.f;
#pragma unroll
    for (int c = 0; c < 5; ++c)
        p += Q[dst * 5 + c] * fmaf(we, We[c], K[src * 5 + c]);
    p *= scale;
    alpha[e] = p;
    atomicMaxF(&mbuf[dst], p);
}

__global__ void edge_msg5(const int* __restrict__ ei, const float* __restrict__ V,
                          const float* __restrict__ We, const float* __restrict__ w,
                          const float* __restrict__ alphaP, const float* __restrict__ sbuf,
                          float* __restrict__ out, int E) {
    int e = blockIdx.x * blockDim.x + threadIdx.x;
    if (e >= E) return;
    int src = ei[e], dst = ei[E + e];
    float we = w[e];
    float a = alphaP[e] / (sbuf[dst] + 1e-16f);
#pragma unroll
    for (int c = 0; c < 5; ++c)
        atomicAdd(&out[dst * 5 + c], fmaf(we, We[c], V[src * 5 + c]) * a);
}

// ---------------- batchnorm + leaky relu ----------------
__global__ void bn_stats(const float* __restrict__ X, float* __restrict__ sum,
                         float* __restrict__ sq, int N) {
    int c = threadIdx.x & 127;
    int r0 = blockIdx.x * 2 + (threadIdx.x >> 7);
    float s = 0.f, q = 0.f;
    for (int r = r0; r < N; r += gridDim.x * 2) {
        float x = X[(size_t)r * 128 + c];
        s += x; q = fmaf(x, x, q);
    }
    atomicAdd(&sum[c], s);
    atomicAdd(&sq[c], q);
}

__global__ void bn_apply(const float* __restrict__ X, const float* __restrict__ sum,
                         const float* __restrict__ sq, const float* __restrict__ g,
                         const float* __restrict__ b, float* __restrict__ H, int N) {
    int idx = blockIdx.x * blockDim.x + threadIdx.x;
    if (idx >= N * 128) return;
    int c = idx & 127;
    float invN = 1.0f / (float)N;
    float mu  = sum[c] * invN;
    float var = sq[c] * invN - mu * mu;
    float y = (X[idx] - mu) * rsqrtf(var + 1e-5f) * g[c] + b[c];
    H[idx] = (y > 0.f) ? y : 0.01f * y;
}

__global__ void fill_f32(float* __restrict__ p, float v, int n) {
    int i = blockIdx.x * blockDim.x + threadIdx.x;
    if (i < n) p[i] = v;
}

// ---------------- host orchestration ----------------
extern "C" void kernel_launch(void* const* d_in, const int* in_sizes, int n_in,
                              void* d_out, int out_size, void* d_ws, size_t ws_size,
                              hipStream_t stream) {
    const float* bf   = (const float*)d_in[1];
    const int*   ei   = (const int*)d_in[3];
    const float* ew   = (const float*)d_in[4];
    const float *Wq1 = (const float*)d_in[5],  *bq1 = (const float*)d_in[6];
    const float *Wk1 = (const float*)d_in[7],  *bk1 = (const float*)d_in[8];
    const float *Wv1 = (const float*)d_in[9],  *bv1 = (const float*)d_in[10];
    const float *We1 = (const float*)d_in[11];
    const float *g1  = (const float*)d_in[12], *b1  = (const float*)d_in[13];
    const float *Wq2 = (const float*)d_in[14], *bq2 = (const float*)d_in[15];
    const float *Wk2 = (const float*)d_in[16], *bk2 = (const float*)d_in[17];
    const float *Wv2 = (const float*)d_in[18], *bv2 = (const float*)d_in[19];
    const float *We2 = (const float*)d_in[20];
    const float *Wskip2 = (const float*)d_in[21], *bskip2 = (const float*)d_in[22];
    const float *g2  = (const float*)d_in[23], *b2  = (const float*)d_in[24];
    const float *Wq3 = (const float*)d_in[25], *bq3 = (const float*)d_in[26];
    const float *Wk3 = (const float*)d_in[27], *bk3 = (const float*)d_in[28];
    const float *Wv3 = (const float*)d_in[29], *bv3 = (const float*)d_in[30];
    const float *We3 = (const float*)d_in[31];
    const float *Wskip3 = (const float*)d_in[32], *bskip3 = (const float*)d_in[33];

    const int N = in_sizes[1] / 15;
    const int E = in_sizes[3] / 2;

    size_t off = 0;
    auto carve = [&](size_t bytes) -> void* {
        void* p = (char*)d_ws + off;
        off += (bytes + 255) & ~(size_t)255;
        return p;
    };
    float* Qb   = (float*)carve((size_t)N * 128 * 4);
    float* Kb   = (float*)carve((size_t)N * 128 * 4);
    float* Vb   = (float*)carve((size_t)N * 128 * 4);
    float* Hb   = (float*)carve((size_t)N * 128 * 4);
    float* OUT  = (float*)carve((size_t)N * 128 * 4);
    float* alphaP = (float*)carve((size_t)E * 4);
    float* a1   = (float*)carve((size_t)E * 4);
    float* a2   = (float*)carve((size_t)E * 4);
    float* mBuf = (float*)carve((size_t)N * 4);
    float* sBuf = (float*)carve((size_t)N * 4);
    float* bnSum = (float*)carve(128 * 4);
    float* bnSq  = (float*)carve(128 * 4);
    __bf16* packedL2 = (__bf16*)carve(4 * 8 * 4 * 32 * 16 * 2);
    __bf16* packedL3 = (__bf16*)carve(4 * 1 * 4 * 32 * 16 * 2);

    const int BLK = 256;
    const int nElem  = N * 128;
    const int gElem  = CDIV(nElem, BLK);
    const int gEdgeW = CDIV(E * 32, BLK);   // warp-per-edge
    const int gEdgeT = CDIV(E, BLK);        // thread-per-edge
    const dim3 gGemm(CDIV(N, 64), 4);       // x: 64-row slabs, y: matrix
    const float scale128 = 0.08838834764831845f;  // 1/sqrt(128)
    const float scale5   = 0.4472135954999579f;   // 1/sqrt(5)

    // ---- pack layer-2 / layer-3 weights into WMMA B fragments ----
    {
        const int t2 = 8 * 4 * 32 * 16, t3 = 1 * 4 * 32 * 16;
        pack_weights_kernel<<<CDIV(t2, BLK), BLK, 0, stream>>>(Wq2,    packedL2 + 0 * t2, 128, 8);
        pack_weights_kernel<<<CDIV(t2, BLK), BLK, 0, stream>>>(Wk2,    packedL2 + 1 * t2, 128, 8);
        pack_weights_kernel<<<CDIV(t2, BLK), BLK, 0, stream>>>(Wv2,    packedL2 + 2 * t2, 128, 8);
        pack_weights_kernel<<<CDIV(t2, BLK), BLK, 0, stream>>>(Wskip2, packedL2 + 3 * t2, 128, 8);
        pack_weights_kernel<<<CDIV(t3, BLK), BLK, 0, stream>>>(Wq3,    packedL3 + 0 * t3, 5, 1);
        pack_weights_kernel<<<CDIV(t3, BLK), BLK, 0, stream>>>(Wk3,    packedL3 + 1 * t3, 5, 1);
        pack_weights_kernel<<<CDIV(t3, BLK), BLK, 0, stream>>>(Wv3,    packedL3 + 2 * t3, 5, 1);
        pack_weights_kernel<<<CDIV(t3, BLK), BLK, 0, stream>>>(Wskip3, packedL3 + 3 * t3, 5, 1);
    }

    // ================= Layer 1 (C=128, no skip) =================
    lin15_qkv<<<gElem, BLK, 0, stream>>>(bf, Wq1, bq1, Wk1, bk1, Wv1, bv1, Qb, Kb, Vb, N);
    fill_f32<<<CDIV(N, BLK), BLK, 0, stream>>>(mBuf, -INFINITY, N);
    fill_f32<<<CDIV(N, BLK), BLK, 0, stream>>>(sBuf, 0.f, N);
    fill_f32<<<gElem, BLK, 0, stream>>>(OUT, 0.f, nElem);
    edge_alpha128<<<gEdgeW, BLK, 0, stream>>>(ei, Qb, Kb, We1, ew, alphaP, mBuf, E, scale128);
    edge_exp<<<gEdgeT, BLK, 0, stream>>>(ei, alphaP, mBuf, sBuf, E);
    edge_msg128<<<gEdgeW, BLK, 0, stream>>>(ei, Vb, We1, ew, alphaP, sBuf, a1, OUT, E);
    fill_f32<<<1, 256, 0, stream>>>(bnSum, 0.f, 128);
    fill_f32<<<1, 256, 0, stream>>>(bnSq, 0.f, 128);
    bn_stats<<<512, BLK, 0, stream>>>(OUT, bnSum, bnSq, N);
    bn_apply<<<gElem, BLK, 0, stream>>>(OUT, bnSum, bnSq, g1, b1, Hb, N);

    // ================= Layer 2 (C=128, skip) =================
    gemm_wmma_v2<<<gGemm, BLK, 0, stream>>>(Hb, packedL2, Qb, Kb, Vb, OUT,
                                            bq2, bk2, bv2, bskip2, N, 128, 128, 8);
    fill_f32<<<CDIV(N, BLK), BLK, 0, stream>>>(mBuf, -INFINITY, N);
    fill_f32<<<CDIV(N, BLK), BLK, 0, stream>>>(sBuf, 0.f, N);
    edge_alpha128<<<gEdgeW, BLK, 0, stream>>>(ei, Qb, Kb, We2, a1, alphaP, mBuf, E, scale128);
    edge_exp<<<gEdgeT, BLK, 0, stream>>>(ei, alphaP, mBuf, sBuf, E);
    edge_msg128<<<gEdgeW, BLK, 0, stream>>>(ei, Vb, We2, a1, alphaP, sBuf, a2, OUT, E);
    fill_f32<<<1, 256, 0, stream>>>(bnSum, 0.f, 128);
    fill_f32<<<1, 256, 0, stream>>>(bnSq, 0.f, 128);
    bn_stats<<<512, BLK, 0, stream>>>(OUT, bnSum, bnSq, N);
    bn_apply<<<gElem, BLK, 0, stream>>>(OUT, bnSum, bnSq, g2, b2, Hb, N);

    // ================= Layer 3 (C=5, skip -> d_out) =================
    float* out3 = (float*)d_out;   // [N,5]; skip GEMM initializes every element
    gemm_wmma_v2<<<gGemm, BLK, 0, stream>>>(Hb, packedL3, Qb, Kb, Vb, out3,
                                            bq3, bk3, bv3, bskip3, N, 5, 5, 1);
    fill_f32<<<CDIV(N, BLK), BLK, 0, stream>>>(mBuf, -INFINITY, N);
    fill_f32<<<CDIV(N, BLK), BLK, 0, stream>>>(sBuf, 0.f, N);
    edge_alpha5<<<gEdgeT, BLK, 0, stream>>>(ei, Qb, Kb, We3, a2, alphaP, mBuf, E, scale5);
    edge_exp<<<gEdgeT, BLK, 0, stream>>>(ei, alphaP, mBuf, sBuf, E);
    edge_msg5<<<gEdgeT, BLK, 0, stream>>>(ei, Vb, We3, a2, alphaP, sBuf, out3, E);

    (void)n_in; (void)out_size; (void)ws_size;
}
